// RowParallelQuantizedLinear_22058952032258
// MI455X (gfx1250) — matrix operand order repaired and checked by reading, not compiled, not run on hardware
//
#include <hip/hip_runtime.h>
#include <hip/hip_bf16.h>
#include <stdint.h>

// ---------------------------------------------------------------------------
// GPTQ 4-bit linear for MI455X (gfx1250, wave32, WMMA):
//   out[M,N] = x[M,K] * (scales*(nib-(z+1)))[K,N] + bias
//   M=8192, K=4096, N=11008, group=128
//
// Compute roofline: 7.4e11 FLOP @ ~5 PFLOPS dense f16 => ~145us floor.
// HBM is only ~22us => VALU work in the K-loop is the enemy.
// => Two-pass: dequant the 45M unique weights ONCE into f16, pre-swizzled
//    into the exact WMMA fragment layout; pre-convert x fp32->f16 likewise.
//    At(67MB)+Wt(90MB) = 157MB < 192MB L2: tile re-reads are L2-served.
//    Main GEMM: zero LDS, zero shuffles, 4x4 register tile, pure
//    global_load_b128 -> v_wmma_f32_16x16x32_f16.
// ---------------------------------------------------------------------------

#define K_DIM 4096
#define N_DIM 11008
#define GROUP 128
#define KT32  (K_DIM / 32)     // 128 K-tiles
#define NT16  (N_DIM / 16)     // 688 N-tiles

typedef __attribute__((ext_vector_type(16))) _Float16 v16h;
typedef __attribute__((ext_vector_type(16))) __bf16   v16bf;
typedef __attribute__((ext_vector_type(8)))  float    v8f;

union FragH  { v16h  v; uint32_t u[8]; uint4 q[2]; };
union FragBF { v16bf v; uint32_t u[8]; uint4 q[2]; };

__device__ __forceinline__ uint32_t pk_f16(float lo, float hi) {
    // v_cvt_pk_f16_f32; builtin returns __fp16 ext_vector(2) -> take raw bits
    return __builtin_bit_cast(uint32_t, __builtin_amdgcn_cvt_pkrtz(lo, hi));
}

// ============================ pass 1a ======================================
// x (fp32, row major) -> At: f16 A-fragments, canonical ISA 16-bit A 16x32
// layout. Tile (mi, ki): lane (khalf=l>>4, m=l&15) holds K runs
// [8*khalf, +8) and [16+8*khalf, +8). 1KB/tile, lane-major, b128-coalesced.
__global__ __launch_bounds__(256)
void cvt_x_tiles(const float* __restrict__ x, uint32_t* __restrict__ At) {
    int gid  = blockIdx.x * 256 + threadIdx.x;       // one thread per (tile,lane)
    int lane = gid & 31;
    int tile = gid >> 5;                             // mi*KT32 + ki
    int mi   = tile / KT32;
    int ki   = tile - mi * KT32;
    int khalf = lane >> 4, m = lane & 15;

    const float* src = x + (size_t)(mi * 16 + m) * K_DIM + ki * 32 + 8 * khalf;
    float4 f0 = *(const float4*)(src);
    float4 f1 = *(const float4*)(src + 4);
    float4 f2 = *(const float4*)(src + 16);
    float4 f3 = *(const float4*)(src + 20);

    uint4 d0, d1;
    d0.x = pk_f16(f0.x, f0.y); d0.y = pk_f16(f0.z, f0.w);
    d0.z = pk_f16(f1.x, f1.y); d0.w = pk_f16(f1.z, f1.w);
    d1.x = pk_f16(f2.x, f2.y); d1.y = pk_f16(f2.z, f2.w);
    d1.z = pk_f16(f3.x, f3.y); d1.w = pk_f16(f3.z, f3.w);

    uint4* dst = (uint4*)(At + ((size_t)tile * 32 + lane) * 8);
    dst[0] = d0;
    dst[1] = d1;
}

// ============================ pass 1b ======================================
// qweight/qzeros/scales -> Wt: f16 B-fragments, canonical B 32x16 layout.
// Tile (ki, ni): lane (khalf, n=l&15) holds K runs [8*khalf, +8) and [16+8*khalf, +8) of column
// ni*16+n  => two qweight dwords, rows 4*ki+khalf and 4*ki+khalf+2.
__global__ __launch_bounds__(256)
void dequant_w_tiles(const int* __restrict__ qw, const int* __restrict__ qz,
                     const float* __restrict__ scales, uint32_t* __restrict__ Wt) {
    int gid  = blockIdx.x * 256 + threadIdx.x;
    int lane = gid & 31;
    int tile = gid >> 5;                             // ki*NT16 + ni
    int ki   = tile / NT16;
    int ni   = tile - ki * NT16;
    int khalf = lane >> 4;
    int n = ni * 16 + (lane & 15);

    int g = ki >> 2;                                 // quant group (K step 32, group 128)
    float s = scales[(size_t)g * N_DIM + n];
    uint32_t zw = (uint32_t)qz[(size_t)g * (N_DIM / 8) + (n >> 3)];
    float nsz = -s * (float)(((zw >> (4 * (n & 7))) & 15u) + 1u);

    int r = ki * 4 + khalf;                          // qweight row (8 K per row)
    uint32_t q0 = (uint32_t)qw[(size_t)r       * N_DIM + n];
    uint32_t q1 = (uint32_t)qw[(size_t)(r + 2) * N_DIM + n];

    uint4 d0, d1;
#define NIB(q, i) ((float)(((q) >> (4 * (i))) & 15u))
    d0.x = pk_f16(__builtin_fmaf(NIB(q0, 0), s, nsz), __builtin_fmaf(NIB(q0, 1), s, nsz));
    d0.y = pk_f16(__builtin_fmaf(NIB(q0, 2), s, nsz), __builtin_fmaf(NIB(q0, 3), s, nsz));
    d0.z = pk_f16(__builtin_fmaf(NIB(q0, 4), s, nsz), __builtin_fmaf(NIB(q0, 5), s, nsz));
    d0.w = pk_f16(__builtin_fmaf(NIB(q0, 6), s, nsz), __builtin_fmaf(NIB(q0, 7), s, nsz));
    d1.x = pk_f16(__builtin_fmaf(NIB(q1, 0), s, nsz), __builtin_fmaf(NIB(q1, 1), s, nsz));
    d1.y = pk_f16(__builtin_fmaf(NIB(q1, 2), s, nsz), __builtin_fmaf(NIB(q1, 3), s, nsz));
    d1.z = pk_f16(__builtin_fmaf(NIB(q1, 4), s, nsz), __builtin_fmaf(NIB(q1, 5), s, nsz));
    d1.w = pk_f16(__builtin_fmaf(NIB(q1, 6), s, nsz), __builtin_fmaf(NIB(q1, 7), s, nsz));
#undef NIB

    uint4* dst = (uint4*)(Wt + ((size_t)tile * 32 + lane) * 8);
    dst[0] = d0;
    dst[1] = d1;
}

// ============================ pass 2: GEMM =================================
// Block 256(M) x 128(N), 8 waves as 4(M)x2(N); wave tile 64x64 = 4x4 frags.
// No LDS, no syncthreads: operands stream from the pre-swizzled f16 buffers
// (both L2-resident) as coalesced b128 loads.
__global__ __launch_bounds__(256)
void gemm_f16_tiles(const uint32_t* __restrict__ At, const uint32_t* __restrict__ Wt,
                    const float* __restrict__ bias, float* __restrict__ out) {
    const int lane  = threadIdx.x & 31;
    const int wave  = threadIdx.x >> 5;
    const int wm    = wave >> 1;           // 0..3
    const int wn    = wave & 1;            // 0..1
    const int khalf = lane >> 4;
    const int l15   = lane & 15;

    const int mblock = blockIdx.y * 256;
    const int nblock = blockIdx.x * 128;
    const int mi0 = (mblock >> 4) + wm * 4;          // first of 4 M-tiles
    const int ni0 = (nblock >> 4) + wn * 4;          // first of 4 N-tiles

    const uint32_t* abase = At + ((size_t)mi0 * KT32 * 32 + lane) * 8;
    const uint32_t* bbase = Wt + ((size_t)ni0 * 32 + lane) * 8;
    const size_t ASTEP = 256;                        // dwords per ki (one tile)
    const size_t BSTEP = (size_t)NT16 * 256;         // dwords per ki row of B tiles
    const size_t AFM   = (size_t)KT32 * 256;         // dwords between M-tiles

    v8f acc[4][4] = {};

    size_t aoff = 0, boff = 0;
#pragma unroll 1
    for (int ki = 0; ki < KT32; ++ki) {
        FragH af[4], bf[4];
#pragma unroll
        for (int fm = 0; fm < 4; ++fm) {
            const uint32_t* p = abase + aoff + (size_t)fm * AFM;
            af[fm].q[0] = *(const uint4*)(p);
            af[fm].q[1] = *(const uint4*)(p + 4);
        }
#pragma unroll
        for (int fn = 0; fn < 4; ++fn) {
            const uint32_t* p = bbase + boff + (size_t)fn * 256;
            bf[fn].q[0] = *(const uint4*)(p);
            bf[fn].q[1] = *(const uint4*)(p + 4);
        }
        // pull tiles 4 K-steps ahead toward the caches (global_prefetch_b8;
        // speculative: OOB prefetches past the end are silently dropped)
        __builtin_prefetch(abase + aoff + 4 * ASTEP, 0, 3);
        __builtin_prefetch(bbase + boff + 4 * BSTEP, 0, 3);

#pragma unroll
        for (int fm = 0; fm < 4; ++fm) {
#pragma unroll
            for (int fn = 0; fn < 4; ++fn) {
                acc[fm][fn] = __builtin_amdgcn_wmma_f32_16x16x32_f16(
                    false, af[fm].v, false, bf[fn].v,
                    (short)0, acc[fm][fn], false, false);
            }
        }
        aoff += ASTEP;
        boff += BSTEP;
    }

    // C layout: n = lane&15, m = r + 8*(lane>>4)
#pragma unroll
    for (int fn = 0; fn < 4; ++fn) {
        int ncol = nblock + wn * 64 + fn * 16 + l15;
        float bv = bias[ncol];
#pragma unroll
        for (int fm = 0; fm < 4; ++fm) {
            size_t base = (size_t)(mblock + wm * 64 + fm * 16 + 8 * khalf) * N_DIM + ncol;
#pragma unroll
            for (int r = 0; r < 8; ++r) {
                out[base + (size_t)r * N_DIM] = acc[fm][fn][r] + bv;
            }
        }
    }
}

// ==================== fallback: fused kernel (round 1) =====================
__device__ __forceinline__ uint32_t bf16_bits(float f) {
    uint32_t b = __builtin_bit_cast(uint32_t, f);
    return (b + 0x7FFFu + ((b >> 16) & 1u)) >> 16;
}
__device__ __forceinline__ uint32_t pk_bf16(float lo, float hi) {
    return bf16_bits(lo) | (bf16_bits(hi) << 16);
}
__device__ __forceinline__ void dq8(uint32_t q, float s, float nsz, uint32_t* d) {
#pragma unroll
    for (int p = 0; p < 4; ++p) {
        float f0 = __builtin_fmaf((float)((q >> (8 * p))     & 15u), s, nsz);
        float f1 = __builtin_fmaf((float)((q >> (8 * p + 4)) & 15u), s, nsz);
        d[p] = pk_bf16(f0, f1);
    }
}

__global__ __launch_bounds__(256)
void gptq_gemm_fused(const float* __restrict__ x, const float* __restrict__ scales,
                     const float* __restrict__ bias, const int* __restrict__ qw,
                     const int* __restrict__ qz, float* __restrict__ out) {
    __shared__ uint32_t As[128 * 20];

    const int tid = threadIdx.x, lane = tid & 31, wave = tid >> 5;
    const int wm = wave >> 1, wn = wave & 1;
    const int lane15 = lane & 15, khalf = lane >> 4;
    const int nblock = blockIdx.x * 128, mblock = blockIdx.y * 128;
    const int ncol0 = nblock + wn * 64 + lane15;

    v8f acc[2][4] = {};
    float sreg[4], nszreg[4];

#pragma unroll 1
    for (int kt = 0; kt < K_DIM; kt += 32) {
        __syncthreads();
#pragma unroll
        for (int i = 0; i < 4; ++i) {
            int idx = tid + i * 256, row = idx >> 3, c4 = idx & 7;
            float4 f = *(const float4*)(x + (size_t)(mblock + row) * K_DIM + kt + c4 * 4);
            uint2 p; p.x = pk_bf16(f.x, f.y); p.y = pk_bf16(f.z, f.w);
            *(uint2*)&As[row * 20 + c4 * 2] = p;
        }
        __syncthreads();

        if ((kt & (GROUP - 1)) == 0) {
            int g = kt >> 7;
#pragma unroll
            for (int fn = 0; fn < 4; ++fn) {
                int ncol = ncol0 + fn * 16;
                float sv = scales[(size_t)g * N_DIM + ncol];
                uint32_t zw = (uint32_t)qz[(size_t)g * (N_DIM / 8) + (ncol >> 3)];
                float zf = (float)(((zw >> (4 * (ncol & 7))) & 15u) + 1u);
                sreg[fn] = sv; nszreg[fn] = -sv * zf;
            }
        }

        FragBF af[2];
#pragma unroll
        for (int fm = 0; fm < 2; ++fm) {
            const uint32_t* asrc = &As[(wm * 32 + fm * 16 + lane15) * 20];
            af[fm].q[0] = *(const uint4*)(asrc + 4 * khalf);
            af[fm].q[1] = *(const uint4*)(asrc + 8 + 4 * khalf);
        }

        const int krow = (kt >> 3) + khalf;
        FragBF bf[4];
#pragma unroll
        for (int fn = 0; fn < 4; ++fn) {
            int ncol = ncol0 + fn * 16;
            uint32_t q0 = (uint32_t)qw[(size_t)krow       * N_DIM + ncol];
            uint32_t q1 = (uint32_t)qw[(size_t)(krow + 2) * N_DIM + ncol];
            dq8(q0, sreg[fn], nszreg[fn], bf[fn].u);
            dq8(q1, sreg[fn], nszreg[fn], bf[fn].u + 4);
        }

#pragma unroll
        for (int fm = 0; fm < 2; ++fm)
#pragma unroll
            for (int fn = 0; fn < 4; ++fn)
                acc[fm][fn] = __builtin_amdgcn_wmma_f32_16x16x32_bf16(
                    false, af[fm].v, false, bf[fn].v, (short)0, acc[fm][fn], false, false);
    }

#pragma unroll
    for (int fn = 0; fn < 4; ++fn) {
        int ncol = ncol0 + fn * 16;
        float bv = bias[ncol];
#pragma unroll
        for (int fm = 0; fm < 2; ++fm) {
            size_t base = (size_t)(mblock + wm * 32 + fm * 16 + 8 * khalf) * N_DIM + ncol;
#pragma unroll
            for (int r = 0; r < 8; ++r)
                out[base + (size_t)r * N_DIM] = acc[fm][fn][r] + bv;
        }
    }
}

// ============================== launcher ===================================
extern "C" void kernel_launch(void* const* d_in, const int* in_sizes, int n_in,
                              void* d_out, int out_size, void* d_ws, size_t ws_size,
                              hipStream_t stream) {
    const float* x      = (const float*)d_in[0];   // (4,2048,4096) fp32
    const float* scales = (const float*)d_in[1];   // (32,11008) fp32
    const float* bias   = (const float*)d_in[2];   // (11008,) fp32
    const int*   qw     = (const int*)d_in[3];     // (512,11008) int32
    const int*   qz     = (const int*)d_in[4];     // (32,1376) int32
    float*       out    = (float*)d_out;           // (4,2048,11008) fp32

    const int M = in_sizes[0] / K_DIM;             // 8192

    const size_t atBytes = (size_t)M * K_DIM * 2;          // 67.1 MB
    const size_t wtBytes = (size_t)K_DIM * N_DIM * 2;      // 90.2 MB

    if (d_ws && ws_size >= atBytes + wtBytes) {
        uint32_t* At = (uint32_t*)d_ws;
        uint32_t* Wt = (uint32_t*)((char*)d_ws + atBytes);

        int aThreads = (M / 16) * KT32 * 32;               // 2,097,152
        int wThreads = KT32 * NT16 * 32;                   // 2,818,048
        cvt_x_tiles    <<<aThreads / 256, 256, 0, stream>>>(x, At);
        dequant_w_tiles<<<wThreads / 256, 256, 0, stream>>>(qw, qz, scales, Wt);

        dim3 grid(N_DIM / 128, M / 256);                   // (86, 32)
        gemm_f16_tiles<<<grid, 256, 0, stream>>>(At, Wt, bias, out);
    } else {
        dim3 grid(N_DIM / 128, M / 128);                   // (86, 64)
        gptq_gemm_fused<<<grid, 256, 0, stream>>>(x, scales, bias, qw, qz, out);
    }
}
